// CAM_Module_17617955848894
// MI455X (gfx1250) — matrix operand order repaired and checked by reading, not compiled, not run on hardware
//
#include <hip/hip_runtime.h>
#include <stdint.h>

// ---------------------------------------------------------------------------
// CDNA5 / gfx1250 implementation.
// GEMMs run on v_wmma_f32_16x16x32_bf16 (wave32), 2x2 register tiling per
// wave (32x32 output block, 4 accumulators), ping-pong double-buffered
// fragment loads (K unrolled by 2, no register rotation -> no v_mov copies).
// Operands pre-packed into the exact per-lane fragment layouts from the
// CDNA5 ISA (CDNA5 ISA):
//   A 16x32 bf16: k_local = (h&7)+((h>>3)<<4)+((lane>>4)<<3), m = lane&15
//   B 32x16 bf16: k_local = (h&7)+((h>>3)<<4)+((lane>>4)<<3),               n = lane&15
//   C/D 16x16 f32: reg r -> M = r + ((lane>>4)<<3), N = lane&15
// qr/kr are rank-1 (einsum contracts only i) so their BN stats come from row
// sums analytically; they are never materialized. k_softmax stages its qk row
// through LDS with GLOBAL_LOAD_ASYNC_TO_LDS_B64 (ASYNCcnt) to overlap the
// rank-1 term computation with the row DMA.
// ---------------------------------------------------------------------------

typedef __attribute__((ext_vector_type(16))) __bf16 v16bf;
typedef __attribute__((ext_vector_type(8)))  float  v8f;

#define LN_EPS 1e-5f

__device__ __forceinline__ unsigned short f2bf(float f) {
  union { float f; unsigned u; } cv; cv.f = f;
  unsigned u = cv.u;
  return (unsigned short)((u + 0x7FFFu + ((u >> 16) & 1u)) >> 16); // RNE
}

// ---------------- operand packing ----------------

// W (1536x512) -> A-pack [Mt=96][Kt=16][32][16]
__global__ void k_pack_w(const float* __restrict__ w, unsigned short* __restrict__ Wp) {
  int id = blockIdx.x * blockDim.x + threadIdx.x;
  if (id >= 96 * 16 * 32) return;
  int lane = id & 31, frag = id >> 5;
  int kt = frag & 15, mt = frag >> 4;
  int m = mt * 16 + (lane & 15);
  unsigned short* d = Wp + ((size_t)frag * 32 + lane) * 16;
#pragma unroll
  for (int h = 0; h < 16; ++h) {
    int k = kt * 32 + (h & 7) + ((h >> 3) << 4) + ((lane >> 4) << 3);
    d[h] = f2bf(w[(size_t)m * 512 + k]);
  }
}

// X (8,512,1024) -> B-pack per b: [b][Nt=64*Kt=16 frags][32][16], B[k=c][n=hw]
__global__ void k_pack_x(const float* __restrict__ x, unsigned short* __restrict__ Xp) {
  int id = blockIdx.x * blockDim.x + threadIdx.x;
  if (id >= 8 * 1024 * 32) return;
  int lane = id & 31, fg = id >> 5;
  int b = fg >> 10, f = fg & 1023;
  int kt = f & 15, nt = f >> 4;
  int n = nt * 16 + (lane & 15);
  unsigned short* d = Xp + ((size_t)fg * 32 + lane) * 16;
#pragma unroll
  for (int h = 0; h < 16; ++h) {
    int k = kt * 32 + (h & 7) + ((h >> 3) << 4) + ((lane >> 4) << 3);
    d[h] = f2bf(x[((size_t)b * 512 + k) * 1024 + n]);
  }
}

// X1 (8,1536,1024, layer-normed) -> Qpack (A), Kpack (B), Vpack (B) per (b,g)
//   q[b,g,c,i] = X1[b, g*192      + c/8, (c%8)*128 + i]
//   k[b,g,j,i] = X1[b, g*192 + 64 + j/8, (j%8)*128 + i]
//   v[b,g,i,j] = X1[b, g*192 +128 + i/8, (i%8)*128 + j]
__global__ void k_pack_qkv(const float* __restrict__ X1,
                           unsigned short* __restrict__ Qp,
                           unsigned short* __restrict__ Kp,
                           unsigned short* __restrict__ Vp) {
  int id = blockIdx.x * blockDim.x + threadIdx.x;
  if (id >= 64 * 384 * 32) return;
  int lane = id & 31, fg = id >> 5;
  int bg = fg / 384, f = fg % 384;
  int b = bg >> 3, g = bg & 7;
  const float* Xb = X1 + (size_t)b * 1536 * 1024;
  unsigned short* d;
  if (f < 128) {                       // Q: A-pack, Mt=32, Kt=4, f = mt*4+kt
    int mt = f >> 2, kt = f & 3;
    int c = mt * 16 + (lane & 15);
    const float* src = Xb + (size_t)(g * 192 + (c >> 3)) * 1024 + ((c & 7) << 7);
    d = Qp + (((size_t)bg * 128 + f) * 32 + lane) * 16;
#pragma unroll
    for (int h = 0; h < 16; ++h) {
      int i = kt * 32 + (h & 7) + ((h >> 3) << 4) + ((lane >> 4) << 3);
      d[h] = f2bf(src[i]);
    }
  } else if (f < 256) {                // K: B-pack, Nt=32, Kt=4, B[i][j]=k[j][i]
    int f2 = f - 128;
    int nt = f2 >> 2, kt = f2 & 3;
    int j = nt * 16 + (lane & 15);
    const float* src = Xb + (size_t)(g * 192 + 64 + (j >> 3)) * 1024 + ((j & 7) << 7);
    d = Kp + (((size_t)bg * 128 + f2) * 32 + lane) * 16;
#pragma unroll
    for (int h = 0; h < 16; ++h) {
      int i = kt * 32 + (h & 7) + ((h >> 3) << 4) + ((lane >> 4) << 3);
      d[h] = f2bf(src[i]);
    }
  } else {                             // V: B-pack, Nt=8, Kt=16, B[i][j]=v[i][j]
    int f3 = f - 256;
    int nt = f3 >> 4, kt = f3 & 15;
    int j = nt * 16 + (lane & 15);
    d = Vp + (((size_t)bg * 128 + f3) * 32 + lane) * 16;
#pragma unroll
    for (int h = 0; h < 16; ++h) {
      int i = kt * 32 + (h & 7) + ((h >> 3) << 4) + ((lane >> 4) << 3);
      d[h] = f2bf(Xb[(size_t)(g * 192 + 128 + (i >> 3)) * 1024 + ((i & 7) << 7) + j]);
    }
  }
}

// ---------------- generic batched bf16 WMMA GEMM, 2x2 wave tiling ----------
// A-pack: [batch][Mt][Kt][32][16], B-pack: [batch][Nt][Kt][32][16]
// One wave computes a 32x32 output block. Kt must be even (ping-pong unroll).
__global__ void k_gemm(const unsigned short* __restrict__ A,
                       const unsigned short* __restrict__ B,
                       float* __restrict__ C,
                       int Mt2, int Nt2, int Kt,
                       unsigned long long aS, unsigned long long bS,
                       unsigned long long cS,
                       int ldc, int transC, const float* scalep) {
  int wid  = (blockIdx.x * blockDim.x + threadIdx.x) >> 5;
  int lane = threadIdx.x & 31;
  int perB = Mt2 * Nt2;
  int batch = wid / perB;
  int r = wid - batch * perB;
  int mt2 = r / Nt2, nt2 = r - mt2 * Nt2;
  size_t fstride = (size_t)Kt * 512;   // ushorts per (tile, all-K) block
  const unsigned short* pa0 = A + (size_t)batch * aS + (size_t)(mt2 * 2) * fstride + lane * 16;
  const unsigned short* pa1 = pa0 + fstride;
  const unsigned short* pb0 = B + (size_t)batch * bS + (size_t)(nt2 * 2) * fstride + lane * 16;
  const unsigned short* pb1 = pb0 + fstride;

  v8f z = {0.f, 0.f, 0.f, 0.f, 0.f, 0.f, 0.f, 0.f};
  v8f acc00 = z, acc01 = z, acc10 = z, acc11 = z;

  // ping buffers: k-step kt
  v16bf a0 = *(const v16bf*)pa0;
  v16bf a1 = *(const v16bf*)pa1;
  v16bf b0 = *(const v16bf*)pb0;
  v16bf b1 = *(const v16bf*)pb1;

  for (int kt = 0; kt < Kt; kt += 2) {
    // pong buffers: k-step kt+1 (Kt even -> always valid)
    size_t o1 = (size_t)(kt + 1) * 512;
    v16bf c0 = *(const v16bf*)(pa0 + o1);
    v16bf c1 = *(const v16bf*)(pa1 + o1);
    v16bf d0 = *(const v16bf*)(pb0 + o1);
    v16bf d1 = *(const v16bf*)(pb1 + o1);
    // speculative prefetch ~2 steps ahead (dropped silently if OOB)
    __builtin_prefetch((const void*)(pa0 + o1 + 1024), 0, 3);
    __builtin_prefetch((const void*)(pb0 + o1 + 1024), 0, 3);

    acc00 = __builtin_amdgcn_wmma_f32_16x16x32_bf16(false, a0, false, b0, (short)0, acc00, false, false);
    acc01 = __builtin_amdgcn_wmma_f32_16x16x32_bf16(false, a0, false, b1, (short)0, acc01, false, false);
    acc10 = __builtin_amdgcn_wmma_f32_16x16x32_bf16(false, a1, false, b0, (short)0, acc10, false, false);
    acc11 = __builtin_amdgcn_wmma_f32_16x16x32_bf16(false, a1, false, b1, (short)0, acc11, false, false);

    if (kt + 2 < Kt) {                 // reload ping buffers for k-step kt+2
      size_t o2 = (size_t)(kt + 2) * 512;
      a0 = *(const v16bf*)(pa0 + o2);
      a1 = *(const v16bf*)(pa1 + o2);
      b0 = *(const v16bf*)(pb0 + o2);
      b1 = *(const v16bf*)(pb1 + o2);
    }

    acc00 = __builtin_amdgcn_wmma_f32_16x16x32_bf16(false, c0, false, d0, (short)0, acc00, false, false);
    acc01 = __builtin_amdgcn_wmma_f32_16x16x32_bf16(false, c0, false, d1, (short)0, acc01, false, false);
    acc10 = __builtin_amdgcn_wmma_f32_16x16x32_bf16(false, c1, false, d0, (short)0, acc10, false, false);
    acc11 = __builtin_amdgcn_wmma_f32_16x16x32_bf16(false, c1, false, d1, (short)0, acc11, false, false);
  }

  float scale = scalep ? scalep[0] : 1.0f;
  float* Cb = C + (size_t)batch * cS;
  int nb = lane & 15;
  int mb = (lane >> 4) << 3;
  v8f accs[4] = {acc00, acc01, acc10, acc11};
#pragma unroll
  for (int ti = 0; ti < 2; ++ti) {
#pragma unroll
    for (int tj = 0; tj < 2; ++tj) {
      v8f acc = accs[ti * 2 + tj];
      int n = (nt2 * 2 + tj) * 16 + nb;
      int mbase = (mt2 * 2 + ti) * 16 + mb;
#pragma unroll
      for (int rr = 0; rr < 8; ++rr) {
        int m = mbase + rr;
        if (!transC) Cb[(size_t)m * ldc + n] = scale * acc[rr];
        else         Cb[(size_t)n * ldc + m] = scale * acc[rr];
      }
    }
  }
}

// ---------------- row LayerNorm (in place) ----------------
__global__ void k_ln_row(float* __restrict__ buf, int len) {
  __shared__ float s1[256], s2[256];
  float* row = buf + (size_t)blockIdx.x * len;
  int tid = threadIdx.x;
  float a = 0.f, b = 0.f;
  for (int i = tid; i < len; i += 256) { float v = row[i]; a += v; b += v * v; }
  s1[tid] = a; s2[tid] = b; __syncthreads();
  for (int s = 128; s > 0; s >>= 1) {
    if (tid < s) { s1[tid] += s1[tid + s]; s2[tid] += s2[tid + s]; }
    __syncthreads();
  }
  float mean = s1[0] / len;
  float var  = s2[0] / len - mean * mean;
  float inv  = rsqrtf(var + LN_EPS);
  for (int i = tid; i < len; i += 256) row[i] = (row[i] - mean) * inv;
}

// ---------------- row sums of q,k (for rank-1 qr/kr) ----------------
__global__ void k_rowsum(const float* __restrict__ X1,
                         float* __restrict__ Qs, float* __restrict__ Ks) {
  int id = blockIdx.x * blockDim.x + threadIdx.x;
  if (id >= 2 * 64 * 512) return;
  int which = id >> 15;
  int r = id & 32767;
  int bg = r >> 9, c = r & 511;
  int b = bg >> 3, g = bg & 7;
  const float* src = X1 + ((size_t)b * 1536 + g * 192 + (which ? 64 : 0) + (c >> 3)) * 1024
                     + ((c & 7) << 7);
  float s = 0.f;
#pragma unroll 4
  for (int i = 0; i < 128; ++i) s += src[i];
  (which ? Ks : Qs)[(size_t)bg * 512 + c] = s;
}

// ---------------- row sums / sums-of-squares of qE, kE ----------------
// qE[c,j] = rel0[c-j+511]; kE[j,c] = rel1[j-c+511]; both reduce to windows.
__global__ void k_rowsumE(const float* __restrict__ rel,
                          float* __restrict__ rsq, float* __restrict__ rssq,
                          float* __restrict__ rsk, float* __restrict__ rssk) {
  int id = blockIdx.x * blockDim.x + threadIdx.x;
  if (id >= 4 * 512) return;
  int t = id >> 9, c = id & 511;
  const float* rr = rel + (t >= 2 ? 1023 : 0);
  float s = 0.f;
  for (int d = 0; d < 512; ++d) {
    float v = rr[c + d];
    s += (t & 1) ? v * v : v;
  }
  if (t == 0) rsq[c] = s;
  else if (t == 1) rssq[c] = s;
  else if (t == 2) rsk[c] = s;
  else rssk[c] = s;
}

// ---------------- BN stats, stage 1: partial sums over qk ----------------
__global__ void k_stats1(const float* __restrict__ QK, float* __restrict__ Part) {
  __shared__ float s1[256], s2[256];
  int g = blockIdx.x >> 6, chunk = blockIdx.x & 63;
  int b = chunk >> 3, c0 = (chunk & 7) << 6;
  const float* base = QK + (((size_t)(b * 8 + g) * 512 + c0) * 512);
  int tid = threadIdx.x;
  float a = 0.f, s = 0.f;
  for (int i = tid; i < 64 * 512; i += 256) { float v = base[i]; a += v; s += v * v; }
  s1[tid] = a; s2[tid] = s; __syncthreads();
  for (int st = 128; st > 0; st >>= 1) {
    if (tid < st) { s1[tid] += s1[tid + st]; s2[tid] += s2[tid + st]; }
    __syncthreads();
  }
  if (tid == 0) { Part[blockIdx.x * 2] = s1[0]; Part[blockIdx.x * 2 + 1] = s2[0]; }
}

// ---------------- BN stats, stage 2: all 24 groups -> (mean, inv_std) -------
__global__ void k_stats2(const float* __restrict__ Part,
                         const float* __restrict__ Qs, const float* __restrict__ Ks,
                         const float* __restrict__ rsq, const float* __restrict__ rssq,
                         const float* __restrict__ rsk, const float* __restrict__ rssk,
                         const float* __restrict__ f_qr, const float* __restrict__ f_kr,
                         float* __restrict__ Stats) {
  __shared__ float s1[256], s2[256];
  int t = blockIdx.x, tid = threadIdx.x;
  float a = 0.f, s = 0.f;
  if (t < 8) {
    for (int i = tid; i < 64; i += 256) {
      a += Part[(t * 64 + i) * 2];
      s += Part[(t * 64 + i) * 2 + 1];
    }
  } else if (t < 16) {
    int g = t - 8;
    for (int c = tid; c < 512; c += 256) {
      float qs = 0.f, qs2 = 0.f;
      for (int b = 0; b < 8; ++b) {
        float v = Qs[(size_t)(b * 8 + g) * 512 + c];
        qs += v; qs2 += v * v;
      }
      a += qs * rsq[c];
      s += qs2 * rssq[c];
    }
  } else {
    int g = t - 16;
    for (int j = tid; j < 512; j += 256) {
      float ks = 0.f, ks2 = 0.f;
      for (int b = 0; b < 8; ++b) {
        float v = Ks[(size_t)(b * 8 + g) * 512 + j];
        ks += v; ks2 += v * v;
      }
      a += ks * rsk[j];
      s += ks2 * rssk[j];
    }
  }
  s1[tid] = a; s2[tid] = s; __syncthreads();
  for (int st = 128; st > 0; st >>= 1) {
    if (tid < st) { s1[tid] += s1[tid + st]; s2[tid] += s2[tid + st]; }
    __syncthreads();
  }
  if (tid == 0) {
    float A = s1[0], S = s2[0];
    if (t >= 8 && t < 16) { float f = f_qr[0]; A *= f; S *= f * f; }
    else if (t >= 16)     { float f = f_kr[0]; A *= f; S *= f * f; }
    const float cnt = 8.0f * 512.0f * 512.0f;
    float mean = A / cnt;
    float var  = S / cnt - mean * mean;
    Stats[t * 2] = mean;
    Stats[t * 2 + 1] = rsqrtf(var + LN_EPS);
  }
}

// ---------------- fused BN + sum + softmax + sim-pack + sve ----------------
__device__ __forceinline__ void simpack_store(unsigned short* Sp, int bg, int c,
                                              int j, float v) {
  int mt = c >> 4, kt = j >> 5, kl = j & 31;
  int hi2 = kl >> 3;
  int lane = ((hi2 & 1) << 4) + (c & 15);
  int h = (kl & 7) + ((hi2 >> 1) << 3);
  Sp[((((size_t)bg * 32 + mt) * 16 + kt) * 32 + lane) * 16 + h] = f2bf(v);
}

__global__ void k_softmax(const float* __restrict__ QK,
                          const float* __restrict__ Qs, const float* __restrict__ Ks,
                          const float* __restrict__ rel,
                          const float* __restrict__ bn_w, const float* __restrict__ bn_b,
                          const float* __restrict__ f_qr, const float* __restrict__ f_kr,
                          const float* __restrict__ f_sve,
                          const float* __restrict__ Stats,
                          unsigned short* __restrict__ Sp,
                          float* __restrict__ SVE) {
  __shared__ float qkbuf[512];
  __shared__ float red[256];
  int rowid = blockIdx.x;
  int bg = rowid >> 9, c = rowid & 511;
  int g = bg & 7;
  int tid = threadIdx.x;
  const float* qkrow = QK + (size_t)rowid * 512;

  // Async-DMA the 2KB qk row into LDS (GLOBAL_LOAD_ASYNC_TO_LDS_B64, GV mode,
  // tracked by ASYNCcnt) while we evaluate the rank-1 qr/kr BN terms.
  {
    unsigned ldsa = (unsigned)(size_t)(&qkbuf[2 * tid]);
    const float* ga = qkrow + 2 * tid;
    asm volatile("global_load_async_to_lds_b64 %0, %1, off"
                 :: "v"(ldsa), "v"(ga) : "memory");
  }

  float m0 = Stats[g * 2],        i0 = Stats[g * 2 + 1];
  float m1 = Stats[(8 + g) * 2],  i1 = Stats[(8 + g) * 2 + 1];
  float m2 = Stats[(16 + g) * 2], i2 = Stats[(16 + g) * 2 + 1];
  float bw0 = bn_w[g],      bb0 = bn_b[g];
  float bw1 = bn_w[8 + g],  bb1 = bn_b[8 + g];
  float bw2 = bn_w[16 + g], bb2 = bn_b[16 + g];
  float qsc = Qs[(size_t)bg * 512 + c] * f_qr[0];
  float fk  = f_kr[0];
  const float* rel0 = rel;
  const float* rel1 = rel + 1023;
  const float* rel2 = rel + 2046;

  int j0 = tid, j1 = tid + 256;
  // rank-1 parts (overlapped with the async row DMA)
  float p0 = (qsc * rel0[c - j0 + 511] - m1) * i1 * bw1 + bb1
           + (fk * Ks[(size_t)bg * 512 + j0] * rel1[j0 - c + 511] - m2) * i2 * bw2 + bb2;
  float p1 = (qsc * rel0[c - j1 + 511] - m1) * i1 * bw1 + bb1
           + (fk * Ks[(size_t)bg * 512 + j1] * rel1[j1 - c + 511] - m2) * i2 * bw2 + bb2;

  asm volatile("s_wait_asynccnt 0x0" ::: "memory");
  __syncthreads();

  float v0 = p0 + (qkbuf[j0] - m0) * i0 * bw0 + bb0;
  float v1 = p1 + (qkbuf[j1] - m0) * i0 * bw0 + bb0;

  // row max
  red[tid] = fmaxf(v0, v1); __syncthreads();
  for (int s = 128; s > 0; s >>= 1) {
    if (tid < s) red[tid] = fmaxf(red[tid], red[tid + s]);
    __syncthreads();
  }
  float rmax = red[0]; __syncthreads();
  float e0 = __expf(v0 - rmax), e1 = __expf(v1 - rmax);
  red[tid] = e0 + e1; __syncthreads();
  for (int s = 128; s > 0; s >>= 1) {
    if (tid < s) red[tid] += red[tid + s];
    __syncthreads();
  }
  float inv = 1.0f / red[0]; __syncthreads();
  float s0 = e0 * inv, s1v = e1 * inv;
  simpack_store(Sp, bg, c, j0, s0);
  simpack_store(Sp, bg, c, j1, s1v);
  // sve[c] = f_sve * sum_j sim[c,j] * vE[c,j], vE[c,j] = rel2[c-j+511]
  red[tid] = s0 * rel2[c - j0 + 511] + s1v * rel2[c - j1 + 511];
  __syncthreads();
  for (int s = 128; s > 0; s >>= 1) {
    if (tid < s) red[tid] += red[tid + s];
    __syncthreads();
  }
  if (tid == 0) SVE[rowid] = f_sve[0] * red[0];
}

// ---------------- final LN stats over t-axis (2048 per (n,c)) ----------------
__global__ void k_ln2stats(const float* __restrict__ SV, const float* __restrict__ SVE,
                           float* __restrict__ Mean2, float* __restrict__ Inv2) {
  __shared__ float s1[256], s2[256];
  int nc = blockIdx.x;
  int n = nc >> 9, c = nc & 511;
  int tid = threadIdx.x;
  float a = 0.f, s = 0.f;
  for (int t = tid; t < 1024; t += 256) {
    int g = t >> 7, p = t & 127;
    float v = SV[(((size_t)(n * 8 + g) * 128) + p) * 512 + c];
    a += v; s += v * v;
  }
  if (tid < 8) {
    float v = SVE[(size_t)(n * 8 + tid) * 512 + c];
    a += 128.0f * v; s += 128.0f * v * v;
  }
  s1[tid] = a; s2[tid] = s; __syncthreads();
  for (int st = 128; st > 0; st >>= 1) {
    if (tid < st) { s1[tid] += s1[tid + st]; s2[tid] += s2[tid + st]; }
    __syncthreads();
  }
  if (tid == 0) {
    float mean = s1[0] / 2048.0f;
    float var  = s2[0] / 2048.0f - mean * mean;
    Mean2[nc] = mean;
    Inv2[nc]  = rsqrtf(var + LN_EPS);
  }
}

// ---------------- final: LN apply + pair-sum + scatter + residual ----------
__global__ void k_final(const float* __restrict__ SV, const float* __restrict__ SVE,
                        const float* __restrict__ Mean2, const float* __restrict__ Inv2,
                        const float* __restrict__ gamma, const float* __restrict__ x,
                        float* __restrict__ out) {
  int id = blockIdx.x * blockDim.x + threadIdx.x;
  if (id >= 8 * 8 * 128 * 512) return;
  int c = id & 511;
  int p = (id >> 9) & 127;
  int g = (id >> 16) & 7;
  int n = id >> 19;
  float sv  = SV[(((size_t)(n * 8 + g) * 128) + p) * 512 + c];
  float sve = SVE[(size_t)(n * 8 + g) * 512 + c];
  float m = Mean2[n * 512 + c], iv = Inv2[n * 512 + c];
  float val = (sv - m) * iv + (sve - m) * iv;
  int cout = g * 64 + (p >> 1);
  int hw = ((p & 1) << 9) | c;
  size_t o = (((size_t)n * 512 + cout) << 10) + hw;
  out[o] = gamma[0] * val + x[o];
}

// ---------------------------------------------------------------------------

extern "C" void kernel_launch(void* const* d_in, const int* in_sizes, int n_in,
                              void* d_out, int out_size, void* d_ws, size_t ws_size,
                              hipStream_t stream) {
  const float* x     = (const float*)d_in[0];
  const float* w_qkv = (const float*)d_in[1];
  const float* rel   = (const float*)d_in[2];
  const float* bn_w  = (const float*)d_in[3];
  const float* bn_b  = (const float*)d_in[4];
  const float* f_qr  = (const float*)d_in[5];
  const float* f_kr  = (const float*)d_in[6];
  const float* f_sv  = (const float*)d_in[7];
  const float* f_sve = (const float*)d_in[8];
  const float* gamma = (const float*)d_in[9];
  float* out = (float*)d_out;
  (void)in_sizes; (void)n_in; (void)out_size; (void)ws_size;

  char* ws = (char*)d_ws;
  size_t off = 0;
  auto alloc = [&](size_t bytes) -> char* {
    char* p = ws + off;
    off += (bytes + 255) & ~(size_t)255;
    return p;
  };
  unsigned short* Wp  = (unsigned short*)alloc(1536ull * 512 * 2);        // 1.5 MB
  unsigned short* Xp  = (unsigned short*)alloc(8ull * 512 * 1024 * 2);    // 8.4 MB
  float*          X1  = (float*)alloc(8ull * 1536 * 1024 * 4);            // 50 MB
  unsigned short* Qp  = (unsigned short*)alloc(64ull * 512 * 128 * 2);    // 8.4 MB
  unsigned short* Kp  = (unsigned short*)alloc(64ull * 512 * 128 * 2);    // 8.4 MB
  unsigned short* Vp  = (unsigned short*)alloc(64ull * 512 * 128 * 2);    // 8.4 MB
  float*          QK  = (float*)alloc(64ull * 512 * 512 * 4);             // 67 MB
  unsigned short* Sp  = (unsigned short*)alloc(64ull * 512 * 512 * 2);    // 34 MB
  float*          SV  = (float*)alloc(64ull * 128 * 512 * 4);             // 17 MB
  float*          SVE = (float*)alloc(64ull * 512 * 4);
  float*          Qs  = (float*)alloc(64ull * 512 * 4);
  float*          Ks  = (float*)alloc(64ull * 512 * 4);
  float*          rsq  = (float*)alloc(512 * 4);
  float*          rssq = (float*)alloc(512 * 4);
  float*          rsk  = (float*)alloc(512 * 4);
  float*          rssk = (float*)alloc(512 * 4);
  float*          Part = (float*)alloc(512 * 2 * 4);
  float*          Stats = (float*)alloc(24 * 2 * 4);
  float*          Mean2 = (float*)alloc(4096 * 4);
  float*          Inv2  = (float*)alloc(4096 * 4);

  // 1) pack GEMM1 operands to bf16 fragments
  k_pack_w<<<192, 256, 0, stream>>>(w_qkv, Wp);
  k_pack_x<<<1024, 256, 0, stream>>>(x, Xp);
  // 2) GEMM1: X1[b] = W(1536x512) * X[b](512x1024); 8 batches, 48x32 wave-tiles
  k_gemm<<<3072, 128, 0, stream>>>(Wp, Xp, X1, 48, 32, 16,
                                   0ull, 524288ull, 1572864ull, 1024, 0, nullptr);
  // 3) row LayerNorm over hw (1024) per (b, o)
  k_ln_row<<<12288, 256, 0, stream>>>(X1, 1024);
  // 4) regather + pack q,k,v fragments; q/k row sums
  k_pack_qkv<<<3072, 256, 0, stream>>>(X1, Qp, Kp, Vp);
  k_rowsum<<<256, 256, 0, stream>>>(X1, Qs, Ks);
  k_rowsumE<<<8, 256, 0, stream>>>(rel, rsq, rssq, rsk, rssk);
  // 5) GEMM2: qk = q * k^T per (b,g); 64 batches, 16x16 wave-tiles
  k_gemm<<<4096, 128, 0, stream>>>(Qp, Kp, QK, 16, 16, 4,
                                   65536ull, 65536ull, 262144ull, 512, 0, nullptr);
  // 6) BN stats (qk direct, qr/kr analytic from row sums)
  k_stats1<<<512, 256, 0, stream>>>(QK, Part);
  k_stats2<<<24, 256, 0, stream>>>(Part, Qs, Ks, rsq, rssq, rsk, rssk,
                                   f_qr, f_kr, Stats);
  // 7) fused BN + group-sum + softmax + bf16 pack of sim + sve (async LDS row)
  k_softmax<<<32768, 256, 0, stream>>>(QK, Qs, Ks, rel, bn_w, bn_b,
                                       f_qr, f_kr, f_sve, Stats, Sp, SVE);
  // 8) GEMM3: sv = (sim * v)^T * f_sv per (b,g); 64 batches, 16x4 wave-tiles
  k_gemm<<<1024, 128, 0, stream>>>(Sp, Vp, SV, 16, 4, 16,
                                   262144ull, 65536ull, 65536ull, 512, 1, f_sv);
  // 9) final LN stats over 2048-wide t axis, apply + scatter + residual
  k_ln2stats<<<4096, 256, 0, stream>>>(SV, SVE, Mean2, Inv2);
  k_final<<<16384, 256, 0, stream>>>(SV, SVE, Mean2, Inv2, gamma, x, out);
}
